// MultiHeadAttention_27195732918332
// MI455X (gfx1250) — matrix operand order repaired and checked by reading, not compiled, not run on hardware
//
#include <hip/hip_runtime.h>
#include <hip/hip_bf16.h>

// ---------------------------------------------------------------------------
// MultiHeadAttention for MI455X (gfx1250, wave32, WMMA bf16 16x16x32).
// B=4, L=2048, D=1024, H=16, d_k=64.
// d_out = out[4,2048,1024] fp32  ||  attn[4,16,2048,2048] fp32
// d_ws  = Qbf(16MB) | Kbf(16MB) | Vt(16MB) | ctx_bf(16MB)   (bf16 tensors)
// ---------------------------------------------------------------------------

#define DMODEL 1024
#define SEQ    2048
#define BATCH  4
#define HEADS  16
#define DKH    64
#define MTOT   (BATCH * SEQ)   // 8192

typedef __attribute__((ext_vector_type(16))) __bf16        bf16x16;
typedef __attribute__((ext_vector_type(8)))  float         f32x8;
typedef __attribute__((ext_vector_type(4)))  float         f32x4;
typedef __attribute__((ext_vector_type(4)))  unsigned int  u32x4;
typedef __attribute__((ext_vector_type(2)))  unsigned int  u32x2;
typedef __attribute__((ext_vector_type(4)))  unsigned int  uint32x4;
typedef __attribute__((ext_vector_type(8)))  int           int32x8;
typedef __attribute__((ext_vector_type(4)))  int           int32x4;

#if __has_builtin(__builtin_amdgcn_tensor_load_to_lds) && \
    __has_builtin(__builtin_amdgcn_s_wait_tensorcnt)
#define USE_TDM 1
#else
#define USE_TDM 0
#endif

union AFrag {
    bf16x16        bf;
    u32x4          q[2];
    unsigned short us[16];
};

__device__ __forceinline__ unsigned short f2bf(float f) {
    union { float f; unsigned u; } v; v.f = f;
    unsigned r = v.u + 0x7FFFu + ((v.u >> 16) & 1u);   // round-to-nearest-even
    return (unsigned short)(r >> 16);
}
__device__ __forceinline__ float bf2f_lo(unsigned u) {
    return __uint_as_float(u << 16);
}
__device__ __forceinline__ float bf2f_hi(unsigned u) {
    return __uint_as_float(u & 0xFFFF0000u);
}

__device__ __forceinline__ f32x8 wmma_bf16(const AFrag& a, const AFrag& b, f32x8 c) {
    // D = A(16x32 bf16) * B(32x16 bf16) + C(16x16 f32)
    return __builtin_amdgcn_wmma_f32_16x16x32_bf16(
        /*neg_a=*/false, a.bf, /*neg_b=*/false, b.bf,
        /*c_mod=*/(short)0, c, /*reuse_a=*/false, /*reuse_b=*/false);
}

// ---------------------------------------------------------------------------
// Kernel 1: fused Q/K/V projections.  X[8192,1024] @ W[1024,1024] + b
// Q,K -> bf16 [B,H,L,64]  ;  V -> bf16 transposed [B,H,64,L]
// ---------------------------------------------------------------------------
__global__ __launch_bounds__(256) void k_proj_qkv(
    const float* __restrict__ q_in, const float* __restrict__ k_in,
    const float* __restrict__ v_in,
    const float* __restrict__ Wq, const float* __restrict__ bq,
    const float* __restrict__ Wk, const float* __restrict__ bk,
    const float* __restrict__ Wv, const float* __restrict__ bv,
    unsigned short* __restrict__ Qbf, unsigned short* __restrict__ Kbf,
    unsigned short* __restrict__ Vt)
{
    const int proj = blockIdx.z;
    const float* X    = (proj == 0) ? q_in : (proj == 1) ? k_in : v_in;
    const float* W    = (proj == 0) ? Wq   : (proj == 1) ? Wk   : Wv;
    const float* bias = (proj == 0) ? bq   : (proj == 1) ? bk   : bv;

    const int m0 = blockIdx.y * 64;
    const int n0 = blockIdx.x * 64;
    const int tid  = threadIdx.x;
    const int w    = tid >> 5;
    const int l    = tid & 31;
    const int half = l >> 4;
    const int ln   = l & 15;
    const int msub = w >> 1;       // 0..3  -> 16-row sub tile
    const int nsub = w & 1;        // 0..1  -> 32-col sub tile (2 WMMA n-tiles)

    __shared__ unsigned short Al[64][40];   // 80B row stride (16B aligned rows)
    __shared__ unsigned short Wl[64][40];   // W^T tile: [n][k]

    f32x8 c0 = {0.f,0.f,0.f,0.f,0.f,0.f,0.f,0.f};
    f32x8 c1 = {0.f,0.f,0.f,0.f,0.f,0.f,0.f,0.f};

    for (int k0 = 0; k0 < DMODEL; k0 += 32) {
        __syncthreads();
        // stage A tile 64x32: float4 loads -> packed bf16 -> b64 LDS stores
#pragma unroll
        for (int t = 0; t < 2; ++t) {
            const int row = (tid >> 3) + 32 * t;
            const int c4  = (tid & 7) * 4;
            const f32x4 x = *(const f32x4*)&X[(long)(m0 + row) * DMODEL + k0 + c4];
            u32x2 p;
            p[0] = (unsigned)f2bf(x[0]) | ((unsigned)f2bf(x[1]) << 16);
            p[1] = (unsigned)f2bf(x[2]) | ((unsigned)f2bf(x[3]) << 16);
            *(u32x2*)&Al[row][c4] = p;
        }
        // stage W tile 32x64 transposed -> Wl[n][k] (coalesced float4 reads)
#pragma unroll
        for (int t = 0; t < 2; ++t) {
            const int kk = (tid >> 4) + 16 * t;
            const int n4 = (tid & 15) * 4;
            const f32x4 wv = *(const f32x4*)&W[(long)(k0 + kk) * DMODEL + n0 + n4];
#pragma unroll
            for (int i = 0; i < 4; ++i) Wl[n4 + i][kk] = f2bf(wv[i]);
        }
        if (k0 + 32 < DMODEL) {   // gfx1250 global_prefetch path
            __builtin_prefetch(&X[(long)(m0 + (tid >> 2)) * DMODEL + k0 + 32], 0, 0);
            __builtin_prefetch(&W[(long)(k0 + 32 + (tid >> 6)) * DMODEL + n0], 0, 0);
        }
        __syncthreads();

        AFrag a, b0, b1;
        // A fragment: row = ln, K = [16h, 16h+16) contiguous (matches B)
        const unsigned short* ar = &Al[16 * msub + ln][0];
        a.q[0] = *(const u32x4*)(ar + 16 * half);
        a.q[1] = *(const u32x4*)(ar + 16 * half + 8);
        // B fragment: col = ln, VGPR v = K pair 2v+16h -> 16 contiguous bf16
        const unsigned short* br0 = &Wl[32 * nsub + ln][16 * half];
        b0.q[0] = *(const u32x4*)(br0);
        b0.q[1] = *(const u32x4*)(br0 + 8);
        const unsigned short* br1 = &Wl[32 * nsub + 16 + ln][16 * half];
        b1.q[0] = *(const u32x4*)(br1);
        b1.q[1] = *(const u32x4*)(br1 + 8);

        c0 = wmma_bf16(a, b0, c0);
        c1 = wmma_bf16(a, b1, c1);
    }

    // epilogue: bias + scatter to attention-friendly layouts
    const int ncol0 = n0 + 32 * nsub + ln;
    const int ncol1 = ncol0 + 16;
    const float bs0 = bias[ncol0];
    const float bs1 = bias[ncol1];
    const int h0 = ncol0 >> 6, d0 = ncol0 & 63;
    const int h1 = ncol1 >> 6, d1 = ncol1 & 63;

#pragma unroll
    for (int r = 0; r < 8; ++r) {
        const int m  = m0 + 16 * msub + r + 8 * half;   // C layout row = r + 8*(lane/16)
        const int bb = m >> 11;        // / SEQ
        const int lq = m & 2047;       // % SEQ
        const unsigned short v0 = f2bf(c0[r] + bs0);
        const unsigned short v1 = f2bf(c1[r] + bs1);
        if (proj == 2) {
            Vt[((long)((bb * HEADS + h0) * DKH + d0)) * SEQ + lq] = v0;
            Vt[((long)((bb * HEADS + h1) * DKH + d1)) * SEQ + lq] = v1;
        } else {
            unsigned short* dst = (proj == 0) ? Qbf : Kbf;
            dst[((long)((bb * HEADS + h0) * SEQ + lq)) * DKH + d0] = v0;
            dst[((long)((bb * HEADS + h1) * SEQ + lq)) * DKH + d1] = v1;
        }
    }
}

// ---------------------------------------------------------------------------
// Kernel 2: attention.  One block = one (b, h, 16-query tile).
// Score/prob row kept in LDS as bf16 (64 KB).
// ---------------------------------------------------------------------------
__global__ __launch_bounds__(256) void k_attn(
    const unsigned short* __restrict__ Qbf, const unsigned short* __restrict__ Kbf,
    const unsigned short* __restrict__ Vt,  float* __restrict__ attn,
    unsigned short* __restrict__ ctx_bf)
{
    __shared__ unsigned short Sb[16][SEQ];   // 64 KB bf16 scores/probs

    const int q0 = blockIdx.x * 16;
    const int h  = blockIdx.y;
    const int b  = blockIdx.z;
    const int bh = b * HEADS + h;
    const int tid  = threadIdx.x;
    const int w    = tid >> 5;
    const int l    = tid & 31;
    const int half = l >> 4;
    const int ln   = l & 15;

    // ---- Q A-fragments (d_k = 64 -> two 16x32 fragments), loaded once ----
    AFrag aq0, aq1;
    {
        const unsigned short* qr = Qbf + ((long)(bh * SEQ + q0 + ln)) * DKH;
        aq0.q[0] = *(const u32x4*)(qr + 16 * half);
        aq0.q[1] = *(const u32x4*)(qr + 16 * half + 8);
        aq1.q[0] = *(const u32x4*)(qr + 32 + 16 * half);
        aq1.q[1] = *(const u32x4*)(qr + 40 + 16 * half);
    }

    // ---- Phase 1: S = scale * Q K^T, waves interleave key tiles ----------
    for (int kt = w; kt < SEQ / 16; kt += 8) {
        const int key0 = kt * 16;
        // B = K^T -> per-lane 16 contiguous bf16 from Kbf[key][dk_slice]
        const unsigned short* kr =
            Kbf + ((long)(bh * SEQ + key0 + ln)) * DKH + 16 * half;
        AFrag b0, b1;
        b0.q[0] = *(const u32x4*)(kr);
        b0.q[1] = *(const u32x4*)(kr + 8);
        b1.q[0] = *(const u32x4*)(kr + 32);
        b1.q[1] = *(const u32x4*)(kr + 40);

        f32x8 c = {0.f,0.f,0.f,0.f,0.f,0.f,0.f,0.f};
        c = wmma_bf16(aq0, b0, c);
        c = wmma_bf16(aq1, b1, c);
#pragma unroll
        for (int r = 0; r < 8; ++r)
            Sb[r + 8 * half][key0 + ln] = f2bf(c[r] * 0.125f);  // 1/sqrt(64)
    }
    __syncthreads();

    // ---- Phase 2: softmax rows (16 thr/row, contiguous 512B chunks) ------
    {
        const int row = tid >> 4;
        const int lr  = tid & 15;
        const int j0  = lr * 128;
        const unsigned short* srow = &Sb[row][j0];

        float m = -1e30f;
        for (int j = 0; j < 128; j += 8) {
            const u32x4 v = *(const u32x4*)(srow + j);
#pragma unroll
            for (int i = 0; i < 4; ++i) {
                m = fmaxf(m, bf2f_lo(v[i]));
                m = fmaxf(m, bf2f_hi(v[i]));
            }
        }
#pragma unroll
        for (int off = 8; off; off >>= 1) m = fmaxf(m, __shfl_xor(m, off, 32));

        float s = 0.f;
        for (int j = 0; j < 128; j += 8) {
            const u32x4 v = *(const u32x4*)(srow + j);
#pragma unroll
            for (int i = 0; i < 4; ++i) {
                s += __expf(bf2f_lo(v[i]) - m);
                s += __expf(bf2f_hi(v[i]) - m);
            }
        }
#pragma unroll
        for (int off = 8; off; off >>= 1) s += __shfl_xor(s, off, 32);
        const float inv = 1.0f / s;

        float* arow = attn + ((long)(bh * SEQ + q0 + row)) * SEQ + j0;
        unsigned short* swr = &Sb[row][j0];
        for (int j = 0; j < 128; j += 8) {
            const u32x4 v = *(const u32x4*)(swr + j);
            f32x4 pa, pb2;
            u32x4 pk;
#pragma unroll
            for (int i = 0; i < 4; ++i) {
                const float plo = __expf(bf2f_lo(v[i]) - m) * inv;
                const float phi = __expf(bf2f_hi(v[i]) - m) * inv;
                if (i < 2) { pa[2*i] = plo; pa[2*i+1] = phi; }
                else       { pb2[2*(i-2)] = plo; pb2[2*(i-2)+1] = phi; }
                pk[i] = (unsigned)f2bf(plo) | ((unsigned)f2bf(phi) << 16);
            }
            *(f32x4*)(arow + j)     = pa;      // normalized attn -> global
            *(f32x4*)(arow + j + 4) = pb2;
            *(u32x4*)(swr + j)      = pk;      // bf16 probs -> LDS for PV
        }
    }
    __syncthreads();

    // ---- Phase 3: ctx = P @ V.  wave -> (key half, dk tile) --------------
    const int half_k = w >> 2;           // which 1024-key half
    const int nt     = w & 3;            // which 16-wide dk tile
    const int n0     = nt * 16;
    f32x8 acc = {0.f,0.f,0.f,0.f,0.f,0.f,0.f,0.f};
    const unsigned short* vrow = Vt + ((long)(bh * DKH + n0 + ln)) * SEQ;
    const unsigned short* prow = &Sb[ln][0];
    for (int ks = 0; ks < 32; ++ks) {
        const int key0 = half_k * 1024 + ks * 32;
        AFrag a, bf;
        // A: contiguous keys [key0+16h, +16) (matches B)
        a.q[0] = *(const u32x4*)(prow + key0 + 16 * half);
        a.q[1] = *(const u32x4*)(prow + key0 + 16 * half + 8);
        bf.q[0] = *(const u32x4*)(vrow + key0 + 16 * half);
        bf.q[1] = *(const u32x4*)(vrow + key0 + 16 * half + 8);
        acc = wmma_bf16(a, bf, acc);
    }
    __syncthreads();                      // all Sb reads done -> overlay scratch
    float* cp = (float*)&Sb[0][0];        // 8 partial 16x16 tiles (8 KB)
#pragma unroll
    for (int r = 0; r < 8; ++r)
        cp[w * 256 + (r + 8 * half) * 16 + ln] = acc[r];
    __syncthreads();

    // combine the two key-halves, emit bf16 ctx in [B, L, H*dk] layout
    for (int idx = tid; idx < 1024; idx += 256) {
        const int nt2 = idx >> 8, rc = idx & 255, row = rc >> 4, col = rc & 15;
        const float v = cp[nt2 * 256 + rc] + cp[(nt2 + 4) * 256 + rc];
        ctx_bf[((long)(b * SEQ + q0 + row)) * DMODEL + h * DKH + nt2 * 16 + col]
            = f2bf(v);
    }
}

// ---------------------------------------------------------------------------
// Kernel 3: output projection.  ctx_bf[8192,1024] @ Wo + bo -> out fp32
// A tile staged by the Tensor Data Mover (TDM) when available.
// ---------------------------------------------------------------------------
__global__ __launch_bounds__(256) void k_proj_out(
    const unsigned short* __restrict__ ctx_bf, const float* __restrict__ Wo,
    const float* __restrict__ bo, float* __restrict__ out)
{
    const int m0 = blockIdx.y * 64;
    const int n0 = blockIdx.x * 64;
    const int tid  = threadIdx.x;
    const int w    = tid >> 5;
    const int l    = tid & 31;
    const int half = l >> 4;
    const int ln   = l & 15;
    const int msub = w >> 1;
    const int nsub = w & 1;

    __shared__ unsigned short Al[64][40];
    __shared__ unsigned short Wl[64][40];

    f32x8 c0 = {0.f,0.f,0.f,0.f,0.f,0.f,0.f,0.f};
    f32x8 c1 = {0.f,0.f,0.f,0.f,0.f,0.f,0.f,0.f};

    for (int k0 = 0; k0 < DMODEL; k0 += 32) {
        __syncthreads();
#if USE_TDM
        // --- TDM: DMA the 64x32 bf16 A tile global->LDS (one wave issues) --
        if (w == 0) {
            const unsigned long long ga =
                (unsigned long long)(size_t)&ctx_bf[(long)m0 * DMODEL + k0];
            const unsigned lds_addr =
                (unsigned)(unsigned long long)(size_t)&Al[0][0];
            uint32x4 g0;
            g0[0] = 1u;                                   // count=1 (valid D#)
            g0[1] = lds_addr;                             // D#.lds_addr
            g0[2] = (unsigned)(ga & 0xFFFFFFFFu);         // global_addr[31:0]
            g0[3] = (unsigned)((ga >> 32) & 0x01FFFFFFu)  // global_addr[56:32]
                    | (2u << 30);                         // type=2 ("image")
            int32x8 g1;
            g1[0] = (int)((1u << 16)      // data_size = 2 bytes
                        | (1u << 20)      // pad_enable
                        | (3u << 22)      // pad_interval: 16 dwords (one row)
                        | (3u << 25));    // pad_amount: 4 dwords (-> 40-u16 rows)
            g1[1] = (int)(1024u << 16);   // tensor_dim0 = 1024 (lo16)
            g1[2] = (int)(8192u << 16);   // dim0 hi | tensor_dim1 = 8192 (lo16)
            g1[3] = (int)(32u << 16);     // dim1 hi | tile_dim0 = 32
            g1[4] = 64;                   // tile_dim1 = 64 | tile_dim2 = 0
            g1[5] = 1024;                 // tensor_dim0_stride[31:0]
            g1[6] = 0;                    // stride hi | tensor_dim1_stride lo
            g1[7] = 0;
            const int32x4 gz  = {0, 0, 0, 0};
            const int32x8 gz8 = {0, 0, 0, 0, 0, 0, 0, 0};
            // 6-arg toolchain form: (g0, g1, g2, g3, g4, cpol)
            __builtin_amdgcn_tensor_load_to_lds(g0, g1, gz, gz, gz8, 0);
            __builtin_amdgcn_s_wait_tensorcnt(0);
        }
#else
        {   // fallback: one b128 copy per thread
            const int row = tid >> 2;
            const int c8  = (tid & 3) * 8;
            *(u32x4*)&Al[row][c8] =
                *(const u32x4*)&ctx_bf[(long)(m0 + row) * DMODEL + k0 + c8];
        }
#endif
        // stage Wo tile transposed
#pragma unroll
        for (int t = 0; t < 2; ++t) {
            const int kk = (tid >> 4) + 16 * t;
            const int n4 = (tid & 15) * 4;
            const f32x4 wv = *(const f32x4*)&Wo[(long)(k0 + kk) * DMODEL + n0 + n4];
#pragma unroll
            for (int i = 0; i < 4; ++i) Wl[n4 + i][kk] = f2bf(wv[i]);
        }
        if (k0 + 32 < DMODEL)
            __builtin_prefetch(&Wo[(long)(k0 + 32 + (tid >> 6)) * DMODEL + n0], 0, 0);
        __syncthreads();

        AFrag a, b0, b1;
        const unsigned short* ar = &Al[16 * msub + ln][0];
        a.q[0] = *(const u32x4*)(ar + 16 * half);
        a.q[1] = *(const u32x4*)(ar + 16 * half + 8);
        const unsigned short* br0 = &Wl[32 * nsub + ln][16 * half];
        b0.q[0] = *(const u32x4*)(br0);
        b0.q[1] = *(const u32x4*)(br0 + 8);
        const unsigned short* br1 = &Wl[32 * nsub + 16 + ln][16 * half];
        b1.q[0] = *(const u32x4*)(br1);
        b1.q[1] = *(const u32x4*)(br1 + 8);

        c0 = wmma_bf16(a, b0, c0);
        c1 = wmma_bf16(a, b1, c1);
    }

    const int ncol0 = n0 + 32 * nsub + ln;
    const int ncol1 = ncol0 + 16;
    const float bs0 = bo[ncol0];
    const float bs1 = bo[ncol1];
#pragma unroll
    for (int r = 0; r < 8; ++r) {
        const int m = m0 + 16 * msub + r + 8 * half;
        out[(long)m * DMODEL + ncol0] = c0[r] + bs0;
        out[(long)m * DMODEL + ncol1] = c1[r] + bs1;
    }
}

// ---------------------------------------------------------------------------
extern "C" void kernel_launch(void* const* d_in, const int* in_sizes, int n_in,
                              void* d_out, int out_size, void* d_ws, size_t ws_size,
                              hipStream_t stream) {
    (void)in_sizes; (void)n_in; (void)out_size; (void)ws_size;

    const float* query = (const float*)d_in[0];
    const float* key_  = (const float*)d_in[1];
    const float* value = (const float*)d_in[2];
    const float* Wq    = (const float*)d_in[3];
    const float* bq    = (const float*)d_in[4];
    const float* Wk    = (const float*)d_in[5];
    const float* bk    = (const float*)d_in[6];
    const float* Wv    = (const float*)d_in[7];
    const float* bv    = (const float*)d_in[8];
    const float* Wo    = (const float*)d_in[9];
    const float* bo    = (const float*)d_in[10];

    float* out  = (float*)d_out;
    float* attn = out + (size_t)MTOT * DMODEL;          // [B,H,L,L]

    const size_t TENS = (size_t)MTOT * DMODEL;          // 8M bf16 = 16 MB each
    unsigned short* Qbf = (unsigned short*)d_ws;
    unsigned short* Kbf = Qbf + TENS;
    unsigned short* Vt  = Kbf + TENS;
    unsigned short* ctx = Vt  + TENS;                   // needs >= 64 MB ws

    k_proj_qkv<<<dim3(DMODEL / 64, MTOT / 64, 3), 256, 0, stream>>>(
        query, key_, value, Wq, bq, Wk, bk, Wv, bv, Qbf, Kbf, Vt);
    k_attn<<<dim3(SEQ / 16, HEADS, BATCH), 256, 0, stream>>>(
        Qbf, Kbf, Vt, attn, ctx);
    k_proj_out<<<dim3(DMODEL / 64, MTOT / 64, 1), 256, 0, stream>>>(
        ctx, Wo, bo, out);
}
